// VQNeighborBasic_26405458936341
// MI455X (gfx1250) — matrix operand order repaired and checked by reading, not compiled, not run on hardware
//
#include <hip/hip_runtime.h>
#include <hip/hip_bf16.h>
#include <stdint.h>

#define N_E   4096
#define NPAD  4112          // 257 * 16 column tiles
#define EDIM  512
#define BB    32
#define TT    512
#define NROW  (BB * TT)     // 16384 rows of X
#define NT    (NPAD / 16)   // 257 column tiles

typedef __attribute__((ext_vector_type(16))) __bf16 v16bf;
typedef __attribute__((ext_vector_type(8)))  float  v8f;
typedef __attribute__((ext_vector_type(4)))  unsigned int u32x4;
typedef __attribute__((ext_vector_type(8)))  int    i32x8;
typedef __attribute__((ext_vector_type(4)))  int    i32x4;

// ---------------------------------------------------------------- convert X
__global__ void k_cvt_key(const float* __restrict__ x, __bf16* __restrict__ y, int n) {
    int i = blockIdx.x * blockDim.x + threadIdx.x;
    int stride = gridDim.x * blockDim.x;
    for (; i < n; i += stride) y[i] = (__bf16)x[i];
}

// ------------------------------------------- convert E (padded) + ||e||^2
__global__ void k_prep_emb(const float* __restrict__ emb, __bf16* __restrict__ ebf,
                           float* __restrict__ esq) {
    const int row = blockIdx.x;      // 0 .. NPAD-1
    const int tid = threadIdx.x;     // 128 threads
    __shared__ float red[128];
    float s = 0.f;
    if (row <= N_E) {
        for (int j = tid; j < EDIM; j += 128) {
            float v = emb[(size_t)row * EDIM + j];
            ebf[(size_t)row * EDIM + j] = (__bf16)v;
            s += v * v;
        }
    } else {
        for (int j = tid; j < EDIM; j += 128)
            ebf[(size_t)row * EDIM + j] = (__bf16)0.f;
    }
    red[tid] = s;
    __syncthreads();
    for (int w = 64; w > 0; w >>= 1) {
        if (tid < w) red[tid] += red[tid + w];
        __syncthreads();
    }
    if (tid == 0) esq[row] = (row <= N_E) ? red[0] : 3.0e38f;
}

// ------------------------- TDM: DMA one 16x512 bf16 tile (16 KB) into LDS
// D# per CDNA5 ISA §8.3/8.4: 2-D tensor, data_size=2B, tile = 512x16,
// row stride 512 elements; groups 2/3 zero (<=2D), workgroup_mask=0.
__device__ __forceinline__ void tdm_load_tile(const __bf16* gptr, unsigned lds_off) {
    unsigned long long ga = (unsigned long long)(uintptr_t)gptr;
    u32x4 g0;
    g0[0] = 1u;                                                   // count=1, user D#
    g0[1] = lds_off;                                              // lds_addr (bytes)
    g0[2] = (unsigned)(ga & 0xFFFFFFFFu);                         // global_addr[31:0]
    g0[3] = (unsigned)((ga >> 32) & 0x01FFFFFFu) | (2u << 30);    // [56:32] | type=2
    i32x8 g1;
    g1[0] = 0x00010000;    // data_size=1 (2 bytes); mask/flags = 0
    g1[1] = 0x02000000;    // tensor_dim0 = 512  (bits[79:48])
    g1[2] = 0x00100000;    // tensor_dim1 = 16   (bits[111:80])
    g1[3] = 0x02000000;    // tile_dim0   = 512  (bits[127:112])
    g1[4] = 16;            // tile_dim1   = 16   (bits[143:128]); tile_dim2 = 0
    g1[5] = 512;           // tensor_dim0_stride = 512 (bits[207:160])
    g1[6] = 0;
    g1[7] = 0;
    i32x4 z4 = {0, 0, 0, 0};
#if defined(__clang_major__) && (__clang_major__ >= 23)
    i32x8 z8 = {0, 0, 0, 0, 0, 0, 0, 0};
    __builtin_amdgcn_tensor_load_to_lds(g0, g1, z4, z4, z8, 0);
#else
    __builtin_amdgcn_tensor_load_to_lds(g0, g1, z4, z4, 0);
#endif
}

// --------------------------- fused bf16-WMMA GEMM + per-row argmin epilogue
// score(m,n) = ||e_n||^2 - 2 * dot(x_m, e_n)   (||x||^2 cancels in comparisons)
// 8 waves/block; each wave owns TWO 16-row M-tiles (A in 256 VGPRs) so every
// B fragment feeds 2 WMMAs. B tiles are TDM-DMA'd into a 2x16KB LDS double
// buffer by wave 0 and shared by all 8 waves. B fragments are double-buffered
// in registers so ds_load latency overlaps the previous k-step's WMMAs.
__global__ __launch_bounds__(256, 2)
void k_gemm_argmin(const __bf16* __restrict__ xb, const __bf16* __restrict__ eb,
                   const float* __restrict__ esq, int* __restrict__ min_idx) {
    __shared__ __bf16 sB[2][16 * EDIM];          // 32 KB double buffer
    const int lane    = threadIdx.x & 31;
    const int wave    = threadIdx.x >> 5;
    const int h       = lane >> 4;               // lane half (0/1)
    const int l16     = lane & 15;
    const int rowbase = blockIdx.x * 256 + wave * 32;
    const bool leader = __builtin_amdgcn_readfirstlane(wave) == 0;  // scalar branch

    // ---- A fragments (16x32 bf16 per k-step): lane half h holds K chunks
    //      [8h..8h+7] and [16+8h..16+8h+7] of its row.
    const __bf16* arow0 = xb + (size_t)(rowbase + l16) * EDIM;
    const __bf16* arow1 = arow0 + (size_t)16 * EDIM;
    v16bf A0[16], A1[16];
#pragma unroll
    for (int k = 0; k < 16; ++k) {
        union { v16bf v; uint4 q[2]; } u;
        const int kb = k * 32 + h * 8;
        u.q[0] = *(const uint4*)(arow0 + kb);
        u.q[1] = *(const uint4*)(arow0 + kb + 16);
        A0[k] = u.v;
        u.q[0] = *(const uint4*)(arow1 + kb);
        u.q[1] = *(const uint4*)(arow1 + kb + 16);
        A1[k] = u.v;
    }

    float best0[8], best1[8];
    int   bidx0[8], bidx1[8];
#pragma unroll
    for (int i = 0; i < 8; ++i) {
        best0[i] = 3.0e38f; bidx0[i] = 0;
        best1[i] = 3.0e38f; bidx1[i] = 0;
    }

    // Prologue: DMA tile 0, wait, publish to all waves.
    if (leader) {
        tdm_load_tile(eb, (unsigned)(uintptr_t)&sB[0][0]);
        __builtin_amdgcn_s_wait_tensorcnt(0);
    }
    __syncthreads();

    for (int nt = 0; nt < NT; ++nt) {
        const int p = nt & 1;
        if (leader && nt + 1 < NT)
            tdm_load_tile(eb + (size_t)(nt + 1) * 16 * EDIM,
                          (unsigned)(uintptr_t)&sB[p ^ 1][0]);

        const __bf16* bbuf = &sB[p][0] + l16 * EDIM;   // column n, contiguous K

        // Register double-buffer for B fragments: issue k+1's loads before
        // consuming k's, so the dscnt wait is partial and overlaps WMMAs.
        union Frag { v16bf v; uint4 q[2]; };
        Frag bu[2];
        bu[0].q[0] = *(const uint4*)(bbuf + h * 8);
        bu[0].q[1] = *(const uint4*)(bbuf + h * 8 + 16);

        v8f c0 = {}, c1 = {};
#pragma unroll
        for (int k = 0; k < 16; ++k) {
            const int cur = k & 1;
            if (k + 1 < 16) {
                const int kb = (k + 1) * 32 + h * 8;
                bu[cur ^ 1].q[0] = *(const uint4*)(bbuf + kb);
                bu[cur ^ 1].q[1] = *(const uint4*)(bbuf + kb + 16);
            }
            c0 = __builtin_amdgcn_wmma_f32_16x16x32_bf16(
                     false, A0[k], false, bu[cur].v, (short)0, c0, false, false);
            c1 = __builtin_amdgcn_wmma_f32_16x16x32_bf16(
                     false, A1[k], false, bu[cur].v, (short)0, c1, false, false);
        }

        const int n = nt * 16 + l16;                   // this lane's column
        const float en = esq[n];
#pragma unroll
        for (int i = 0; i < 8; ++i) {
            // c[i] <-> row M = i + 8h, col N = l16 of the tile (C layout)
            float s0 = en - 2.0f * c0[i];
            bool t0 = s0 < best0[i];
            best0[i] = t0 ? s0 : best0[i];
            bidx0[i] = t0 ? n : bidx0[i];
            float s1 = en - 2.0f * c1[i];
            bool t1 = s1 < best1[i];
            best1[i] = t1 ? s1 : best1[i];
            bidx1[i] = t1 ? n : bidx1[i];
        }

        if (leader) __builtin_amdgcn_s_wait_tensorcnt(0);
        __syncthreads();
    }

    // Cross-lane argmin within each 16-lane half (xor masks stay in-half).
#pragma unroll
    for (int i = 0; i < 8; ++i) {
        float b0 = best0[i]; int i0 = bidx0[i];
        float b1 = best1[i]; int i1 = bidx1[i];
#pragma unroll
        for (int m = 8; m >= 1; m >>= 1) {
            float ob0 = __shfl_xor(b0, m, 32);
            int   oi0 = __shfl_xor(i0, m, 32);
            bool t0 = (ob0 < b0) || (ob0 == b0 && oi0 < i0);  // first-occurrence tie
            b0 = t0 ? ob0 : b0;
            i0 = t0 ? oi0 : i0;
            float ob1 = __shfl_xor(b1, m, 32);
            int   oi1 = __shfl_xor(i1, m, 32);
            bool t1 = (ob1 < b1) || (ob1 == b1 && oi1 < i1);
            b1 = t1 ? ob1 : b1;
            i1 = t1 ? oi1 : i1;
        }
        bidx0[i] = i0;
        bidx1[i] = i1;
    }
    if (l16 == 0) {
        const int mb0 = rowbase + h * 8;
        const int mb1 = rowbase + 16 + h * 8;
#pragma unroll
        for (int i = 0; i < 8; ++i) {
            min_idx[mb0 + i] = bidx0[i];
            min_idx[mb1 + i] = bidx1[i];
        }
    }
}

// -------------------------------- sequential neighbor scan, one block per b
__global__ void k_scan(const float* __restrict__ x, const float* __restrict__ emb,
                       const unsigned char* __restrict__ mask,
                       const int* __restrict__ min_idx,
                       int* __restrict__ enc, float* __restrict__ encf,
                       int* __restrict__ range) {
    const int b   = blockIdx.x;
    const int tid = threadIdx.x;   // 256 threads
    __shared__ float r1[256], r2[256];
    __shared__ int   s_ind;

    if (tid == 0) {
        int i0 = min_idx[b * TT];
        if (i0 > N_E - 1) i0 = N_E - 1;
        if (mask[b]) i0 = 0;
        s_ind = i0;
    }
    __syncthreads();
    int ind = s_ind;
    if (tid == 0) { enc[b * TT] = ind; encf[b * TT] = (float)ind; }
    int bmin = ind, bmax = ind;

    for (int t = 1; t < TT; ++t) {
        const float* xr = x + (size_t)(b * TT + t) * EDIM;
        if (t + 1 < TT) __builtin_prefetch((const void*)(xr + EDIM), 0, 1);
        int indn = ind + 1;
        if (indn > N_E - 1) indn = N_E - 1;
        const float* eh = emb + (size_t)ind  * EDIM;
        const float* en = emb + (size_t)indn * EDIM;
        float s1 = 0.f, s2 = 0.f;
        for (int j = tid; j < EDIM; j += 256) {
            float xv = xr[j];
            float d1 = xv - eh[j]; s1 += d1 * d1;
            float d2 = xv - en[j]; s2 += d2 * d2;
        }
        r1[tid] = s1; r2[tid] = s2;
        __syncthreads();
        for (int w = 128; w > 0; w >>= 1) {
            if (tid < w) { r1[tid] += r1[tid + w]; r2[tid] += r2[tid + w]; }
            __syncthreads();
        }
        if (tid == 0) {
            int nind = (r1[0] <= r2[0]) ? ind : indn;   // tie keeps ind
            s_ind = nind;
            enc[b * TT + t]  = nind;
            encf[b * TT + t] = (float)nind;
        }
        __syncthreads();
        ind = s_ind;
        bmin = (ind < bmin) ? ind : bmin;
        bmax = (ind > bmax) ? ind : bmax;
    }
    if (tid == 0) range[b] = bmax - bmin;
}

// -------------------- gathers + losses + straight-through forward output
__global__ void k_loss(const float* __restrict__ x, const float* __restrict__ emb,
                       const int* __restrict__ enc, const int* __restrict__ min_idx,
                       float* __restrict__ key_hard, float* __restrict__ loss_here,
                       float* __restrict__ loss_next) {
    const int row = blockIdx.x;     // 0 .. NROW-1
    const int tid = threadIdx.x;    // 128 threads
    __shared__ float r1[128], r2[128], r3[128];

    int ind  = enc[row];
    int indn = ind + 1;
    if (indn > N_E - 1) indn = N_E - 1;
    int mi = min_idx[row];

    const float* xr = x   + (size_t)row  * EDIM;
    const float* eh = emb + (size_t)ind  * EDIM;
    const float* en = emb + (size_t)indn * EDIM;
    const float* em = emb + (size_t)mi   * EDIM;

    float s1 = 0.f, s2 = 0.f, s3 = 0.f;
    for (int j = tid; j < EDIM; j += 128) {
        float xv = xr[j];
        float vh = eh[j];
        key_hard[(size_t)row * EDIM + j] = vh;   // x + sg(e - x) == e forward
        float d1 = xv - vh;    s1 += d1 * d1;
        float d2 = xv - en[j]; s2 += d2 * d2;
        float d3 = xv - em[j]; s3 += d3 * d3;
    }
    r1[tid] = s1; r2[tid] = s2; r3[tid] = s3;
    __syncthreads();
    for (int w = 64; w > 0; w >>= 1) {
        if (tid < w) { r1[tid] += r1[tid+w]; r2[tid] += r2[tid+w]; r3[tid] += r3[tid+w]; }
        __syncthreads();
    }
    if (tid == 0) {
        float lh = 1.2f * r1[0];   // (1 + LEGACY) * sum((x-e)^2), forward value
        float ln = 1.2f * r2[0];
        float lm = 1.2f * r3[0];
        loss_here[row] = lh - ((lm < lh) ? lm : 0.f);
        loss_next[row] = ln - ((lm < ln) ? lm : 0.f);
    }
}

// ---------------------------------------------------------------- scalar v
__global__ void k_vmax(const int* __restrict__ range, float* __restrict__ vout) {
    if (threadIdx.x == 0 && blockIdx.x == 0) {
        int m = range[0];
        for (int b = 1; b < BB; ++b) m = (range[b] > m) ? range[b] : m;
        *vout = (float)m;
    }
}

extern "C" void kernel_launch(void* const* d_in, const int* in_sizes, int n_in,
                              void* d_out, int out_size, void* d_ws, size_t ws_size,
                              hipStream_t stream) {
    const float*         key_soft = (const float*)d_in[0];
    const float*         emb      = (const float*)d_in[1];
    const unsigned char* mask     = (const unsigned char*)d_in[2];

    char* ws = (char*)d_ws;
    size_t off = 0;
    __bf16* key_bf = (__bf16*)(ws + off); off += (size_t)NROW * EDIM * 2;
    __bf16* emb_bf = (__bf16*)(ws + off); off += (size_t)NPAD * EDIM * 2;
    float*  esq    = (float*)(ws + off);  off += (size_t)NPAD * 4;
    int*    midx   = (int*)(ws + off);    off += (size_t)NROW * 4;
    int*    enc    = (int*)(ws + off);    off += (size_t)NROW * 4;
    int*    range  = (int*)(ws + off);    off += (size_t)BB * 4;

    float* out        = (float*)d_out;
    float* o_key_hard = out;                              // [B,T,D]
    float* o_enc      = out + (size_t)NROW * EDIM;        // [B,T] as float
    float* o_v        = o_enc + NROW;                     // scalar
    float* o_lh       = o_v + 1;                          // [B,T]
    float* o_ln       = o_lh + NROW;                      // [B,T]

    k_cvt_key<<<2048, 256, 0, stream>>>(key_soft, key_bf, NROW * EDIM);
    k_prep_emb<<<NPAD, 128, 0, stream>>>(emb, emb_bf, esq);
    k_gemm_argmin<<<64, 256, 0, stream>>>(key_bf, emb_bf, esq, midx);
    k_scan<<<BB, 256, 0, stream>>>(key_soft, emb, mask, midx, enc, o_enc, range);
    k_loss<<<NROW, 128, 0, stream>>>(key_soft, emb, enc, midx, o_key_hard, o_lh, o_ln);
    k_vmax<<<1, 32, 0, stream>>>(range, o_v);
}
